// CustomMultiHeadSelfAttention_35742717837546
// MI455X (gfx1250) — matrix operand order repaired and checked by reading, not compiled, not run on hardware
//
#include <hip/hip_runtime.h>
#include <hip/hip_bf16.h>

// Problem constants (from reference)
#define EMB   1024
#define N3    3072
#define SEQ   2048
#define NB    4
#define NH    16
#define DKH   64
#define NROW  (NB * SEQ)   // 8192 tokens

typedef __attribute__((ext_vector_type(16))) __bf16 v16bf;
typedef __attribute__((ext_vector_type(8)))  float  v8f;
typedef __attribute__((ext_vector_type(8)))  unsigned int v8u;

// ---- gfx1250 async global->LDS copy (ASYNCcnt-tracked), with fallback ----
#if defined(__has_builtin)
#if __has_builtin(__builtin_amdgcn_global_load_async_to_lds_b128)
#define HAS_ASYNC_LDS 1
#endif
#endif
#ifndef HAS_ASYNC_LDS
#define HAS_ASYNC_LDS 0
#endif

// The builtin takes pointers to 16-byte int vectors (GCC vector_size style),
// qualified with target address spaces 1 (global) and 3 (LDS).
typedef int v4i_ld __attribute__((vector_size(4 * sizeof(int))));

// Flat->AS casts via integers: for LDS the low 32 bits of a flat address are
// the LDS offset (ISA 10.2 aperture mapping); global flat == global.
#define TO_GLB(p) ((__attribute__((address_space(1))) v4i_ld*)(unsigned long long)(p))
#define TO_LDS(p) ((__attribute__((address_space(3))) v4i_ld*)(unsigned int)(unsigned long long)(p))

__device__ __forceinline__ void async_copy16(const void* g, void* l) {
#if HAS_ASYNC_LDS
  __builtin_amdgcn_global_load_async_to_lds_b128(TO_GLB(g), TO_LDS(l), 0, 0);
#else
  *(uint4*)l = *(const uint4*)g;
#endif
}

__device__ __forceinline__ void wait_async0() {
#if HAS_ASYNC_LDS
#if __has_builtin(__builtin_amdgcn_s_wait_asynccnt)
  __builtin_amdgcn_s_wait_asynccnt(0);
#else
  asm volatile("s_wait_asynccnt 0" ::: "memory");
#endif
#endif
}

__device__ __forceinline__ v8f wmma_bf16(v16bf a, v16bf b, v8f c) {
  // D = A(16x32 bf16) * B(32x16 bf16) + C(16x16 f32)
  return __builtin_amdgcn_wmma_f32_16x16x32_bf16(false, a, false, b,
                                                 (short)0, c, false, false);
}

__device__ __forceinline__ v8f vzero() {
  v8f z = {0.f, 0.f, 0.f, 0.f, 0.f, 0.f, 0.f, 0.f};
  return z;
}

__device__ __forceinline__ v16bf frag_from2(uint4 lo, uint4 hi) {
  v8u u;
  u[0] = lo.x; u[1] = lo.y; u[2] = lo.z; u[3] = lo.w;
  u[4] = hi.x; u[5] = hi.y; u[6] = hi.z; u[7] = hi.w;
  return __builtin_bit_cast(v16bf, u);
}

// ---- WMMA fragment gathers (CDNA5 ISA 7.12.2 layouts, wave32) ----
// Each loader is exactly two 16-byte loads per lane (ds_load_b128 /
// global_load_b128); lda/ldb must be a multiple of 8 elements and the
// base 16B-aligned.

// A fragment 16x32 bf16 from row-major [m][k] storage.
// Lane l: m = l%16, h = l/16. VGPR i holds K = (i>=4?16:0) + 8h + (i%4)*2.
// Per lane: row[8h .. 8h+7] ++ row[8h+16 .. 8h+23]: two contiguous 16B.
__device__ __forceinline__ v16bf load_frag_a(const __bf16* p, int lda) {
  const int lane = threadIdx.x & 31;
  const int m = lane & 15;
  const int h = lane >> 4;
  const __bf16* row = p + m * lda + (h << 3);
  const uint4 lo = *(const uint4*)(row);
  const uint4 hi = *(const uint4*)(row + 16);
  return frag_from2(lo, hi);
}

// B fragment 32x16 bf16 from [n][k] storage (ldb elems per n-row).
// Lane l: n = l%16, h = l/16. VGPR i holds K = (i>=4?16:0) + 8h + (i%4)*2:
// col[8h .. 8h+7] ++ col[8h+16 .. 8h+23] -> two contiguous 16B loads.
__device__ __forceinline__ v16bf load_frag_b_t(const __bf16* p, int ldb) {
  const int lane = threadIdx.x & 31;
  const int n = lane & 15;
  const int h = lane >> 4;
  const __bf16* col = p + n * ldb + (h << 3);
  const uint4 lo = *(const uint4*)(col);
  const uint4 hi = *(const uint4*)(col + 16);
  return frag_from2(lo, hi);
}

// Pack 4 f32 -> 4 bf16 (8 bytes) for LDS staging stores.
__device__ __forceinline__ uint2 pack4_bf16(float4 v) {
  union { __bf16 h[4]; uint2 u; } pk;
  pk.h[0] = (__bf16)v.x; pk.h[1] = (__bf16)v.y;
  pk.h[2] = (__bf16)v.z; pk.h[3] = (__bf16)v.w;
  return pk.u;
}

// =====================================================================
// Kernel 1: qkv = x @ W_qkv + b_qkv, scattered to bf16 Q,K [B,H,T,64]
// and V transposed [B,H,64,T]. Block tile 128x128, 8 waves of 32x64.
// =====================================================================
__global__ __launch_bounds__(256)
void qkv_gemm_kernel(const float* __restrict__ x,
                     const float* __restrict__ Wqkv,
                     const float* __restrict__ bqkv,
                     __bf16* __restrict__ Q,
                     __bf16* __restrict__ K,
                     __bf16* __restrict__ Vt) {
  __shared__ alignas(16) __bf16 Al[128][40];  // [m][k]
  __shared__ alignas(16) __bf16 Bl[128][40];  // [n][k] (transposed on store)

  const int tid = threadIdx.x;
  const int w   = tid >> 5;
  const int wm  = w >> 1;           // 0..3 -> 32-row slab
  const int wn  = w & 1;            // 0..1 -> 64-col slab
  const int lane = tid & 31, ln = lane & 15, hh = lane >> 4;
  const int gm0 = blockIdx.y * 128;
  const int gn0 = blockIdx.x * 128;

  v8f acc[2][4];
#pragma unroll
  for (int i = 0; i < 2; ++i)
#pragma unroll
    for (int j = 0; j < 4; ++j) acc[i][j] = vzero();

  for (int k0 = 0; k0 < EMB; k0 += 32) {
    __syncthreads();
    // Stage A: 128x32 of x, f32 -> bf16, contiguous 8B stores.
#pragma unroll
    for (int it = 0; it < 4; ++it) {
      const int a = tid + it * 256;           // 0..1023
      const int r = a >> 3, cg = (a & 7) * 4; // row 0..127, col 0,4,..28
      const float4 v = *(const float4*)(x + (size_t)(gm0 + r) * EMB + k0 + cg);
      *(uint2*)&Al[r][cg] = pack4_bf16(v);
    }
    // Stage B: 32x128 of W_qkv, f32 -> bf16, transposed into [n][k].
#pragma unroll
    for (int it = 0; it < 4; ++it) {
      const int a = tid + it * 256;
      const int r = a >> 5, cg = (a & 31) * 4; // k 0..31, n 0,4,..124
      const float4 v = *(const float4*)(Wqkv + (size_t)(k0 + r) * N3 + gn0 + cg);
      Bl[cg + 0][r] = (__bf16)v.x;
      Bl[cg + 1][r] = (__bf16)v.y;
      Bl[cg + 2][r] = (__bf16)v.z;
      Bl[cg + 3][r] = (__bf16)v.w;
    }
    __syncthreads();

    const v16bf a0 = load_frag_a(&Al[wm * 32][0], 40);
    const v16bf a1 = load_frag_a(&Al[wm * 32 + 16][0], 40);
#pragma unroll
    for (int j = 0; j < 4; ++j) {
      const v16bf bb = load_frag_b_t(&Bl[wn * 64 + j * 16][0], 40);
      acc[0][j] = wmma_bf16(a0, bb, acc[0][j]);
      acc[1][j] = wmma_bf16(a1, bb, acc[1][j]);
    }
  }

  // Epilogue: bias + scatter to head-major bf16 buffers.
#pragma unroll
  for (int i = 0; i < 2; ++i) {
#pragma unroll
    for (int j = 0; j < 4; ++j) {
      const int gn    = gn0 + wn * 64 + j * 16;  // tile col base (mult of 16)
      const int which = gn / EMB;                // 0=Q 1=K 2=V (uniform per tile)
      const int c0    = gn % EMB;
      const int h     = c0 >> 6;                 // head
      const int dk    = (c0 & 63) + ln;
      const float bias = bqkv[gn + ln];
#pragma unroll
      for (int r = 0; r < 8; ++r) {
        const int m    = gm0 + wm * 32 + i * 16 + hh * 8 + r;  // token row
        const int bidx = m >> 11;                // / SEQ
        const int t    = m & (SEQ - 1);
        const float val = acc[i][j][r] + bias;
        if (which == 0)
          Q[(((size_t)bidx * NH + h) * SEQ + t) * DKH + dk] = (__bf16)val;
        else if (which == 1)
          K[(((size_t)bidx * NH + h) * SEQ + t) * DKH + dk] = (__bf16)val;
        else
          Vt[(((size_t)bidx * NH + h) * DKH + dk) * SEQ + t] = (__bf16)val;
      }
    }
  }
}

// =====================================================================
// Kernel 2: flash attention. One (b,h) per blockIdx.y, 128 query rows
// per block (8 waves x 16 rows). Double-buffered K/V tiles streamed
// into LDS with global_load_async_to_lds_b128 (ASYNCcnt) while the
// current tile's WMMAs + online softmax run. Scores never touch HBM.
// =====================================================================
__global__ __launch_bounds__(256)
void flash_attn_kernel(const __bf16* __restrict__ Q,
                       const __bf16* __restrict__ K,
                       const __bf16* __restrict__ Vt,
                       __bf16* __restrict__ AO) {
  __shared__ alignas(16) __bf16 Kl[2][32][72];     // [s][dk] (matches K layout)
  __shared__ alignas(16) __bf16 Vl[2][64][40];     // [dk][s] (matches Vt layout)
  __shared__ alignas(16) __bf16 Pl[8][16][40];     // per-wave P repack [m][k]

  const int tid = threadIdx.x;
  const int w   = tid >> 5;
  const int lane = tid & 31, ln = lane & 15, hh = lane >> 4;
  const int bh = blockIdx.y;
  const int bb = bh >> 4, h = bh & 15;
  const int q0 = blockIdx.x * 128 + w * 16;   // this wave's query row base

  const __bf16* Qg = Q  + (((size_t)bb * NH + h) * SEQ + q0) * DKH;
  const __bf16* Kg = K  + (((size_t)bb * NH + h) * SEQ) * DKH;
  const __bf16* Vg = Vt + (((size_t)bb * NH + h) * DKH) * SEQ;

  // Q tile 16x64 held in registers as two A fragments.
  const v16bf qf0 = load_frag_a(Qg, DKH);
  const v16bf qf1 = load_frag_a(Qg + 32, DKH);

  // Per-thread staging slots: 16B of K-tile and 16B of V-tile each.
  const int idx = tid * 8;                   // 0..2040
  const int sk = idx >> 6, kd = idx & 63;    // K tile: 32 x 64
  const int dv = idx >> 5, sv = idx & 31;    // V tile: 64 x 32

  float mrow[8], lrow[8];
  v8f o[4];
#pragma unroll
  for (int r = 0; r < 8; ++r) { mrow[r] = -1e30f; lrow[r] = 0.f; }
#pragma unroll
  for (int j = 0; j < 4; ++j) o[j] = vzero();

  // Prologue: kick off tile 0.
  async_copy16(Kg + (size_t)sk * DKH + kd, &Kl[0][sk][kd]);
  async_copy16(Vg + (size_t)dv * SEQ + sv, &Vl[0][dv][sv]);

  for (int s0 = 0; s0 < SEQ; s0 += 32) {
    const int cur = (s0 >> 5) & 1;
    wait_async0();        // this wave's copies for `cur` have landed in LDS
    __syncthreads();      // => all waves' copies landed; prior compute done

    if (s0 + 32 < SEQ) {  // stream next tile into the other buffer
      async_copy16(Kg + (size_t)(s0 + 32 + sk) * DKH + kd, &Kl[cur ^ 1][sk][kd]);
      async_copy16(Vg + (size_t)dv * SEQ + s0 + 32 + sv,   &Vl[cur ^ 1][dv][sv]);
    }

    // S = Q (16x64) @ K^T (64x32) -> two 16x16 f32 tiles
    v8f sa0 = vzero(), sa1 = vzero();
    v16bf fb;
    fb = load_frag_b_t(&Kl[cur][0][0],   72); sa0 = wmma_bf16(qf0, fb, sa0);
    fb = load_frag_b_t(&Kl[cur][0][32],  72); sa0 = wmma_bf16(qf1, fb, sa0);
    fb = load_frag_b_t(&Kl[cur][16][0],  72); sa1 = wmma_bf16(qf0, fb, sa1);
    fb = load_frag_b_t(&Kl[cur][16][32], 72); sa1 = wmma_bf16(qf1, fb, sa1);

    // Online softmax across the 32 keys (rows spread over 16 lanes).
#pragma unroll
    for (int r = 0; r < 8; ++r) {
      float x0 = sa0[r] * 0.125f;    // 1/sqrt(64)
      float x1 = sa1[r] * 0.125f;
      float mx = fmaxf(x0, x1);
#pragma unroll
      for (int off = 1; off < 16; off <<= 1)
        mx = fmaxf(mx, __shfl_xor(mx, off, 16));
      const float mnew = fmaxf(mrow[r], mx);
      const float corr = __expf(mrow[r] - mnew);
      const float p0 = __expf(x0 - mnew);
      const float p1 = __expf(x1 - mnew);
      float ps = p0 + p1;
#pragma unroll
      for (int off = 1; off < 16; off <<= 1)
        ps += __shfl_xor(ps, off, 16);
      lrow[r] = lrow[r] * corr + ps;
      mrow[r] = mnew;
#pragma unroll
      for (int j = 0; j < 4; ++j) o[j][r] = o[j][r] * corr;
      // Repack P (C layout) -> row-major [m][k] for the A fragment.
      const int m = hh * 8 + r;
      Pl[w][m][ln]      = (__bf16)p0;
      Pl[w][m][ln + 16] = (__bf16)p1;
    }

    // O += P (16x32) @ V (32x64)
    const v16bf pa = load_frag_a(&Pl[w][0][0], 40);
#pragma unroll
    for (int j = 0; j < 4; ++j) {
      const v16bf vb = load_frag_b_t(&Vl[cur][j * 16][0], 40);
      o[j] = wmma_bf16(pa, vb, o[j]);
    }
  }

  // Normalize and write attn output in [B,T,H*dk] = [B,T,D] order (bf16).
#pragma unroll
  for (int r = 0; r < 8; ++r) {
    const int m = hh * 8 + r;
    const float rl = 1.f / lrow[r];
    const size_t row = (size_t)bb * SEQ + q0 + m;
#pragma unroll
    for (int j = 0; j < 4; ++j) {
      AO[row * EMB + h * DKH + j * 16 + ln] = (__bf16)(o[j][r] * rl);
    }
  }
}

// =====================================================================
// Kernel 3: out = AO @ W_out + b_out; res = x + out; LayerNorm(res).
// Each block owns 16 full rows (LN reduction stays in-block);
// 8 waves x 128 cols, 8 WMMA accumulators per wave.
// =====================================================================
__global__ __launch_bounds__(256)
void proj_ln_kernel(const __bf16* __restrict__ AO,
                    const float* __restrict__ Wout,
                    const float* __restrict__ bout,
                    const float* __restrict__ x,
                    const float* __restrict__ gamma,
                    const float* __restrict__ beta,
                    float* __restrict__ out) {
  __shared__ alignas(16) __bf16 Al[16][40];    // [m][k]
  __shared__ alignas(16) __bf16 Bl[1024][40];  // [n][k] (transposed on store)
  __shared__ float red[2][16][8];              // [sum|sumsq][row][wave]

  const int tid = threadIdx.x;
  const int w   = tid >> 5;
  const int lane = tid & 31, ln = lane & 15, hh = lane >> 4;
  const int row0 = blockIdx.x * 16;

  v8f acc[8];
#pragma unroll
  for (int j = 0; j < 8; ++j) acc[j] = vzero();

  for (int k0 = 0; k0 < EMB; k0 += 32) {
    __syncthreads();
    if (tid < 64) {                    // stage A: 16x32 bf16 (already bf16)
      const int r = tid >> 2, c = (tid & 3) * 8;
      *(uint4*)&Al[r][c] =
          *(const uint4*)(AO + (size_t)(row0 + r) * EMB + k0 + c);
    }
    for (int it = 0; it < 32; ++it) {  // stage B: 32x1024 f32 -> bf16, [n][k]
      const int a = tid + it * 256;            // 0..8191
      const int r = a >> 8, cg = (a & 255) * 4;
      const float4 v = *(const float4*)(Wout + (size_t)(k0 + r) * EMB + cg);
      Bl[cg + 0][r] = (__bf16)v.x;
      Bl[cg + 1][r] = (__bf16)v.y;
      Bl[cg + 2][r] = (__bf16)v.z;
      Bl[cg + 3][r] = (__bf16)v.w;
    }
    __syncthreads();

    const v16bf af = load_frag_a(&Al[0][0], 40);
#pragma unroll
    for (int j = 0; j < 8; ++j) {
      const v16bf bf = load_frag_b_t(&Bl[w * 128 + j * 16][0], 40);
      acc[j] = wmma_bf16(af, bf, acc[j]);
    }
  }

  // Bias + residual
#pragma unroll
  for (int j = 0; j < 8; ++j) {
    const int col = w * 128 + j * 16 + ln;
    const float bias = bout[col];
#pragma unroll
    for (int r = 0; r < 8; ++r) {
      const int m = hh * 8 + r;
      acc[j][r] = acc[j][r] + bias + x[(size_t)(row0 + m) * EMB + col];
    }
  }

  // LayerNorm: per-row mean/var over 1024 cols (lanes x tiles x waves).
  float s[8], sq[8];
#pragma unroll
  for (int r = 0; r < 8; ++r) {
    float a = 0.f, b2 = 0.f;
#pragma unroll
    for (int j = 0; j < 8; ++j) { a += acc[j][r]; b2 += acc[j][r] * acc[j][r]; }
#pragma unroll
    for (int off = 1; off < 16; off <<= 1) {
      a  += __shfl_xor(a,  off, 16);
      b2 += __shfl_xor(b2, off, 16);
    }
    s[r] = a; sq[r] = b2;
  }
  if (ln == 0) {
#pragma unroll
    for (int r = 0; r < 8; ++r) {
      red[0][hh * 8 + r][w] = s[r];
      red[1][hh * 8 + r][w] = sq[r];
    }
  }
  __syncthreads();
#pragma unroll
  for (int r = 0; r < 8; ++r) {
    const int m = hh * 8 + r;
    float a = 0.f, b2 = 0.f;
#pragma unroll
    for (int ww = 0; ww < 8; ++ww) { a += red[0][m][ww]; b2 += red[1][m][ww]; }
    const float mu   = a * (1.f / EMB);
    const float var  = b2 * (1.f / EMB) - mu * mu;
    const float rstd = rsqrtf(var + 1e-5f);
#pragma unroll
    for (int j = 0; j < 8; ++j) {
      const int col = w * 128 + j * 16 + ln;
      out[(size_t)(row0 + m) * EMB + col] =
          (acc[j][r] - mu) * rstd * gamma[col] + beta[col];
    }
  }
}

// =====================================================================
extern "C" void kernel_launch(void* const* d_in, const int* in_sizes, int n_in,
                              void* d_out, int out_size, void* d_ws, size_t ws_size,
                              hipStream_t stream) {
  (void)in_sizes; (void)n_in; (void)out_size; (void)ws_size;
  const float* x     = (const float*)d_in[0];
  const float* Wqkv  = (const float*)d_in[1];
  const float* bqkv  = (const float*)d_in[2];
  const float* Wout  = (const float*)d_in[3];
  const float* bout  = (const float*)d_in[4];
  const float* gamma = (const float*)d_in[5];
  const float* beta  = (const float*)d_in[6];
  float* out = (float*)d_out;

  // Workspace layout (bf16): Q | K | Vt | AO, 16 MB each (64 MB total).
  const size_t QSZ = (size_t)NB * NH * SEQ * DKH;   // 8,388,608 elems
  __bf16* Q  = (__bf16*)d_ws;
  __bf16* K  = Q + QSZ;
  __bf16* Vt = K + QSZ;
  __bf16* AO = Vt + QSZ;

  qkv_gemm_kernel<<<dim3(N3 / 128, NROW / 128), 256, 0, stream>>>(
      x, Wqkv, bqkv, Q, K, Vt);
  flash_attn_kernel<<<dim3(SEQ / 128, NB * NH), 256, 0, stream>>>(
      Q, K, Vt, AO);
  proj_ln_kernel<<<dim3(NROW / 16), 256, 0, stream>>>(
      AO, Wout, bout, x, gamma, beta, out);
}
